// ObjectBranch_drg_67774583931070
// MI455X (gfx1250) — matrix operand order repaired and checked by reading, not compiled, not run on hardware
//
#include <hip/hip_runtime.h>
#include <hip/hip_bf16.h>

// ---------------------------------------------------------------------------
// MI455X (gfx1250) implementation of the ObjectBranch reference:
//   ROI-align -> relu(A@W1+b1) -> relu(H1@W2+b2) -> mean over 7x7 -> pair avg
// GEMMs run on v_wmma_f32_16x16x32_f16 (f16 in, f32 accumulate) with
// double-buffered LDS tiles; A tiles staged with GLOBAL_LOAD_ASYNC_TO_LDS
// (ASYNCcnt) issued via inline asm (the ROCm builtin's signature is
// toolchain-dependent; the mnemonic is not).
// ---------------------------------------------------------------------------

typedef __attribute__((ext_vector_type(16))) _Float16 v16h;
typedef __attribute__((ext_vector_type(8)))  float    v8f;

// Flip to 0 to fall back to VGPR-staged LDS tiles (round-1 codegen).
#define USE_ASYNC_LDS 1

#if USE_ASYNC_LDS
// Copy 16 bytes global -> LDS without touching VGPR data lanes.
// VDST operand = LDS byte address (low 32 bits of the generic pointer,
// per ISA 10.2: generic->LDS uses addr[31:0]); VADDR = 64-bit global addr.
__device__ __forceinline__ void async_copy_b128(void* lds_dst, const void* gsrc) {
    asm volatile("global_load_async_to_lds_b128 %0, %1, off"
                 :
                 : "v"((unsigned)(unsigned long long)lds_dst),
                   "v"((unsigned long long)gsrc)
                 : "memory");
}
#endif

__device__ __forceinline__ void wait_asynccnt0() {
#if defined(__has_builtin) && __has_builtin(__builtin_amdgcn_s_wait_asynccnt)
    __builtin_amdgcn_s_wait_asynccnt(0);
#else
    asm volatile("s_wait_asynccnt 0x0" ::: "memory");
#endif
}

namespace cfg {
constexpr int B     = 32;
constexpr int NOBJ  = 16;
constexpr int NREL  = 64;
constexpr int C     = 1024;
constexpr int H     = 50;
constexpr int W     = 50;
constexpr int POOL  = 7;
constexpr int SR    = 2;
constexpr int D0    = 1024;   // roi feature dim (== C)
constexpr int D1    = 1024;
constexpr int D2    = 512;
constexpr int NROI  = B * NOBJ;          // 512
constexpr int NBIN  = POOL * POOL;       // 49
constexpr int MROWS = NROI * NBIN;       // 25088 = 196 * 128
}

// ---------------------------------------------------------------------------
// Bilinear axis helper (matches reference _interp_axis semantics exactly)
// ---------------------------------------------------------------------------
struct Axis { int lo, hi; float w0, w1, v; };

__device__ __forceinline__ Axis interp_axis(float coord, int size) {
    Axis a;
    a.v = (coord >= -1.0f && coord <= (float)size) ? 1.0f : 0.0f;
    float c = fmaxf(coord, 0.0f);
    int low = (int)floorf(c);
    bool edge = low >= size - 1;
    low = min(low, size - 1);
    a.lo = low;
    a.hi = min(low + 1, size - 1);
    float l = edge ? 0.0f : (c - (float)low);
    a.w0 = 1.0f - l;
    a.w1 = l;
    return a;
}

// ---------------------------------------------------------------------------
// Kernel 1: weight conversion f32 -> f16 (done once per launch into ws)
// ---------------------------------------------------------------------------
__global__ void convert_weights_kernel(const float* __restrict__ W1,
                                       const float* __restrict__ W2,
                                       _Float16* __restrict__ W1h,
                                       _Float16* __restrict__ W2h) {
    const size_t stride = (size_t)gridDim.x * blockDim.x;
    size_t i = (size_t)blockIdx.x * blockDim.x + threadIdx.x;
    for (size_t j = i; j < (size_t)cfg::D0 * cfg::D1; j += stride)
        W1h[j] = (_Float16)W1[j];
    for (size_t j = i; j < (size_t)cfg::D1 * cfg::D2; j += stride)
        W2h[j] = (_Float16)W2[j];
}

// ---------------------------------------------------------------------------
// Kernel 2: ROI align. One block per (roi, bin); 256 threads sweep channels.
// Emits A (MROWS x 1024) in f16, row-major, ready for WMMA GEMM1.
// ---------------------------------------------------------------------------
__global__ __launch_bounds__(256)
void roi_align_kernel(const float* __restrict__ fmap,
                      const float* __restrict__ bboxes,
                      const int* __restrict__ frame_w,
                      const int* __restrict__ frame_h,
                      _Float16* __restrict__ A) {
    const int blk = blockIdx.x;            // roi*49 + bin
    const int roi = blk / cfg::NBIN;
    const int bin = blk - roi * cfg::NBIN;
    const int py  = bin / cfg::POOL;
    const int px  = bin - py * cfg::POOL;
    const int b   = roi / cfg::NOBJ;

    // Reference swaps: im_height, im_width = (frame_width, frame_height)
    const float im_h = (float)frame_w[0];
    const float im_w = (float)frame_h[0];
    const float wsc  = (float)cfg::W / im_w;
    const float hsc  = (float)cfg::H / im_h;

    const float* bb = bboxes + (size_t)roi * 4;
    const float x1 = bb[0] * wsc, y1 = bb[1] * hsc;
    const float x2 = bb[2] * wsc, y2 = bb[3] * hsc;
    const float roi_w = fmaxf(x2 - x1, 1.0f);
    const float roi_h = fmaxf(y2 - y1, 1.0f);
    const float bw = roi_w / (float)cfg::POOL;
    const float bh = roi_h / (float)cfg::POOL;

    // Precompute the 4 sample points (2x2) of this bin: indices + fused weights
    int   idx[4][4];
    float wgt[4][4];
    #pragma unroll
    for (int sy = 0; sy < cfg::SR; ++sy) {
        const float gy = (float)py + ((float)sy + 0.5f) / (float)cfg::SR;
        const Axis ay = interp_axis(y1 + bh * gy, cfg::H);
        #pragma unroll
        for (int sx = 0; sx < cfg::SR; ++sx) {
            const float gx = (float)px + ((float)sx + 0.5f) / (float)cfg::SR;
            const Axis ax = interp_axis(x1 + bw * gx, cfg::W);
            const int s = sy * cfg::SR + sx;
            const float v = ay.v * ax.v;
            idx[s][0] = ay.lo * cfg::W + ax.lo;
            idx[s][1] = ay.lo * cfg::W + ax.hi;
            idx[s][2] = ay.hi * cfg::W + ax.lo;
            idx[s][3] = ay.hi * cfg::W + ax.hi;
            wgt[s][0] = ay.w0 * ax.w0 * v;
            wgt[s][1] = ay.w0 * ax.w1 * v;
            wgt[s][2] = ay.w1 * ax.w0 * v;
            wgt[s][3] = ay.w1 * ax.w1 * v;
        }
    }

    const size_t bbase = (size_t)b * cfg::C * (cfg::H * cfg::W);
    for (int c = threadIdx.x; c < cfg::C; c += 256) {
        const float* f = fmap + bbase + (size_t)c * (cfg::H * cfg::W);
        float acc = 0.0f;
        #pragma unroll
        for (int s = 0; s < 4; ++s) {
            acc += wgt[s][0] * f[idx[s][0]] + wgt[s][1] * f[idx[s][1]] +
                   wgt[s][2] * f[idx[s][2]] + wgt[s][3] * f[idx[s][3]];
        }
        A[(size_t)blk * cfg::D0 + c] = (_Float16)(acc * 0.25f);
    }
}

// ---------------------------------------------------------------------------
// WMMA GEMM: out = relu(A @ B + bias), A (M x K) f16 row-major,
// B (K x N) f16 row-major. 256 threads = 8 waves, block tile 128x128,
// K-step 32, double-buffered LDS. Wave (wm,wn) = 2x4 grid, each wave owns a
// 64x32 sub-tile = 4x2 accumulators of v_wmma_f32_16x16x32_f16.
//
// Staging per K-step:
//   A tile (128x32 f16 = 8 KB): 512 16-byte chunks; each thread issues two
//     GLOBAL_LOAD_ASYNC_TO_LDS_B128 (ASYNCcnt), bypassing VGPRs. Row pitch
//     40 halves = 80 B keeps every chunk 16-B aligned and spreads rows
//     across LDS banks without systematic conflicts.
//   B tile (32x128) must be transposed into Bs[n][k] so each fragment is two
//     contiguous 8-half runs -> staged via registers (rb[]), which also forms
//     the second pipeline stage.
// One barrier per K-step; next tile's loads are issued before the current
// tile's fragment reads + WMMAs so the copy overlaps the math.
//
// Fragment lane layouts per ISA 7.12.2:
//   A frag (16x32 MxK): lane<16 row=l16 holds K 0..7 & 16..23,
//                       lane>=16 holds K 8..15 & 24..31.
//   B frag (32x16 KxN): lane<16 col=l16 holds K 0..7 & 16..23, lane>=16 K 8..15 & 24..31.
//   C/D: lane<16 col=l16 rows base+v, lane>=16 rows base+8+v.
// ---------------------------------------------------------------------------
template <int N, int K, bool OUT_HALF>
__global__ __launch_bounds__(256)
void wmma_gemm_relu_kernel(const _Float16* __restrict__ Ag,
                           const _Float16* __restrict__ Bg,
                           const float* __restrict__ bias,
                           void* __restrict__ out) {
    __shared__ __align__(16) _Float16 As[2][128][40];   // [buf][m][k]
    __shared__ __align__(16) _Float16 Bs[2][128][40];   // [buf][n][k]

    const int m0   = blockIdx.x * 128;
    const int n0   = blockIdx.y * 128;
    const int tid  = threadIdx.x;
    const int lane = tid & 31;
    const int wave = tid >> 5;     // 0..7
    const int wm   = wave >> 2;    // 0..1 -> 64 rows each
    const int wn   = wave & 3;     // 0..3 -> 32 cols each
    const int l16  = lane & 15;
    const int hi16 = lane >> 4;    // 0 | 1

    v8f acc[4][2] = {};
    _Float16 rb[16];
#if !USE_ASYNC_LDS
    uint4 ra[2];
#endif

    // ---- prologue: stage K-step 0 into buffer 0 -------------------------
    #pragma unroll
    for (int i = 0; i < 16; ++i) {
        const int lin = tid + 256 * i;
        rb[i] = Bg[(size_t)(lin >> 7) * N + (n0 + (lin & 127))];
    }
#if USE_ASYNC_LDS
    #pragma unroll
    for (int i = 0; i < 2; ++i) {
        const int q = tid + 256 * i;            // 16-B chunk id, 0..511
        const int r = q >> 2, kc = q & 3;
        async_copy_b128(&As[0][r][kc * 8],
                        Ag + (size_t)(m0 + r) * K + kc * 8);
    }
#else
    #pragma unroll
    for (int i = 0; i < 2; ++i) {
        const int q = tid + 256 * i;
        const int r = q >> 2, kc = q & 3;
        ra[i] = *(const uint4*)(Ag + (size_t)(m0 + r) * K + kc * 8);
    }
    #pragma unroll
    for (int i = 0; i < 2; ++i) {
        const int q = tid + 256 * i;
        const int r = q >> 2, kc = q & 3;
        *(uint4*)(&As[0][r][kc * 8]) = ra[i];
    }
#endif
    #pragma unroll
    for (int i = 0; i < 16; ++i) {
        const int lin = tid + 256 * i;
        Bs[0][lin & 127][lin >> 7] = rb[i];
    }
#if USE_ASYNC_LDS
    wait_asynccnt0();
#endif
    __syncthreads();

    // ---- main loop: one barrier per K-step, next tile overlapped --------
    for (int k0 = 0; k0 < K; k0 += 32) {
        const int  cur      = (k0 >> 5) & 1;
        const int  nxt      = cur ^ 1;
        const bool has_next = (k0 + 32) < K;

        if (has_next) {
            #pragma unroll
            for (int i = 0; i < 16; ++i) {
                const int lin = tid + 256 * i;
                rb[i] = Bg[(size_t)(k0 + 32 + (lin >> 7)) * N + (n0 + (lin & 127))];
            }
#if USE_ASYNC_LDS
            #pragma unroll
            for (int i = 0; i < 2; ++i) {
                const int q = tid + 256 * i;
                const int r = q >> 2, kc = q & 3;
                async_copy_b128(&As[nxt][r][kc * 8],
                                Ag + (size_t)(m0 + r) * K + (k0 + 32) + kc * 8);
            }
#else
            #pragma unroll
            for (int i = 0; i < 2; ++i) {
                const int q = tid + 256 * i;
                const int r = q >> 2, kc = q & 3;
                ra[i] = *(const uint4*)(Ag + (size_t)(m0 + r) * K + (k0 + 32) + kc * 8);
            }
#endif
        }

        // B fragments: two contiguous 8-half runs at K = kb and kb+16 (match A)
        v16h bf[2];
        #pragma unroll
        for (int j = 0; j < 2; ++j) {
            const int n = wn * 32 + j * 16 + l16;
            const int kb = hi16 * 8;
            #pragma unroll
            for (int e = 0; e < 16; ++e) bf[j][e] = Bs[cur][n][kb + (e & 7) + ((e >> 3) << 4)];
        }
        // A fragments: two contiguous 8-half runs at K = kb and kb+16
        v16h af[4];
        #pragma unroll
        for (int i = 0; i < 4; ++i) {
            const int r = wm * 64 + i * 16 + l16;
            const int kb = hi16 * 8;
            #pragma unroll
            for (int e = 0; e < 8; ++e) af[i][e]     = As[cur][r][kb + e];
            #pragma unroll
            for (int e = 0; e < 8; ++e) af[i][e + 8] = As[cur][r][kb + 16 + e];
        }

        #pragma unroll
        for (int i = 0; i < 4; ++i)
            #pragma unroll
            for (int j = 0; j < 2; ++j)
                acc[i][j] = __builtin_amdgcn_wmma_f32_16x16x32_f16(
                    false, af[i], false, bf[j], (short)0, acc[i][j],
                    false, false);

        if (has_next) {
#if !USE_ASYNC_LDS
            #pragma unroll
            for (int i = 0; i < 2; ++i) {
                const int q = tid + 256 * i;
                const int r = q >> 2, kc = q & 3;
                *(uint4*)(&As[nxt][r][kc * 8]) = ra[i];
            }
#endif
            #pragma unroll
            for (int i = 0; i < 16; ++i) {
                const int lin = tid + 256 * i;
                Bs[nxt][lin & 127][lin >> 7] = rb[i];
            }
#if USE_ASYNC_LDS
            wait_asynccnt0();
#endif
        }
        __syncthreads();
    }

    // ---- epilogue -------------------------------------------------------
    const int rofs = hi16 * 8;
    #pragma unroll
    for (int i = 0; i < 4; ++i) {
        #pragma unroll
        for (int j = 0; j < 2; ++j) {
            const int col = n0 + wn * 32 + j * 16 + l16;
            const float bv = bias[col];
            #pragma unroll
            for (int v = 0; v < 8; ++v) {
                const int row = m0 + wm * 64 + i * 16 + rofs + v;
                float x = fmaxf(acc[i][j][v] + bv, 0.0f);
                if (OUT_HALF)
                    ((_Float16*)out)[(size_t)row * N + col] = (_Float16)x;
                else
                    ((float*)out)[(size_t)row * N + col] = x;
            }
        }
    }
}

// ---------------------------------------------------------------------------
// Kernel 5: deterministic mean over the 49 bins of each ROI -> feat (512x512)
// ---------------------------------------------------------------------------
__global__ __launch_bounds__(256)
void bin_mean_kernel(const float* __restrict__ H2, float* __restrict__ feat) {
    const int roi = blockIdx.x;
    for (int c = threadIdx.x; c < cfg::D2; c += 256) {
        float s = 0.0f;
        #pragma unroll 7
        for (int r = 0; r < cfg::NBIN; ++r)
            s += H2[(size_t)(roi * cfg::NBIN + r) * cfg::D2 + c];
        feat[(size_t)roi * cfg::D2 + c] = s * (1.0f / (float)cfg::NBIN);
    }
}

// ---------------------------------------------------------------------------
// Kernel 6: pair gather: out[b,r] = 0.5*(feat[off_b+p0] + feat[off_b+p1])
// ---------------------------------------------------------------------------
__global__ __launch_bounds__(256)
void pair_gather_kernel(const int* __restrict__ num_obj,
                        const int* __restrict__ obj_pairs,
                        const float* __restrict__ feat,
                        float* __restrict__ out) {
    const int row = blockIdx.x;               // 0 .. B*NREL-1
    const int b = row / cfg::NREL;
    const int r = row - b * cfg::NREL;
    int off = 0;
    for (int i = 0; i < b; ++i) off += num_obj[i];
    const int p0 = off + obj_pairs[((size_t)b * cfg::NREL + r) * 2 + 0];
    const int p1 = off + obj_pairs[((size_t)b * cfg::NREL + r) * 2 + 1];
    const float* f0 = feat + (size_t)p0 * cfg::D2;
    const float* f1 = feat + (size_t)p1 * cfg::D2;
    for (int c = threadIdx.x; c < cfg::D2; c += 256)
        out[(size_t)row * cfg::D2 + c] = 0.5f * (f0[c] + f1[c]);
}

// ---------------------------------------------------------------------------
// Launch. Workspace layout (bytes):
//   [0)                A f16  (MROWS x 1024)           = 51,380,224
//   [offA)             W1 f16 (1024x1024)              =  2,097,152
//   [offW1)            W2 f16 (1024x512)               =  1,048,576
//   [offW2)            H1 f16 (MROWS x 1024)           = 51,380,224
//   [offH1)            feat f32 (512x512)              =  1,048,576
//   H2 f32 (MROWS x 512) aliases the A region (A is dead after GEMM1).
// Total ~102 MB.
// ---------------------------------------------------------------------------
extern "C" void kernel_launch(void* const* d_in, const int* in_sizes, int n_in,
                              void* d_out, int out_size, void* d_ws,
                              size_t ws_size, hipStream_t stream) {
    (void)in_sizes; (void)n_in; (void)out_size; (void)ws_size;
    using namespace cfg;

    const float* fmap      = (const float*)d_in[0];
    const float* bboxes    = (const float*)d_in[1];
    const int*   num_obj   = (const int*)d_in[2];
    const int*   obj_pairs = (const int*)d_in[3];
    /* d_in[4] = num_rels (unused: NREL is the padded per-batch count) */
    const int*   frame_w   = (const int*)d_in[5];
    const int*   frame_h   = (const int*)d_in[6];
    const float* W1        = (const float*)d_in[7];
    const float* b1        = (const float*)d_in[8];
    const float* W2        = (const float*)d_in[9];
    const float* b2        = (const float*)d_in[10];
    float* out = (float*)d_out;

    char* ws = (char*)d_ws;
    _Float16* A    = (_Float16*)ws;
    const size_t offA  = (size_t)MROWS * D0 * sizeof(_Float16);
    _Float16* W1h  = (_Float16*)(ws + offA);
    const size_t offW1 = offA + (size_t)D0 * D1 * sizeof(_Float16);
    _Float16* W2h  = (_Float16*)(ws + offW1);
    const size_t offW2 = offW1 + (size_t)D1 * D2 * sizeof(_Float16);
    _Float16* H1   = (_Float16*)(ws + offW2);
    const size_t offH1 = offW2 + (size_t)MROWS * D1 * sizeof(_Float16);
    float*    feat = (float*)(ws + offH1);
    float*    H2   = (float*)ws;   // aliases A (dead after GEMM1)

    convert_weights_kernel<<<2048, 256, 0, stream>>>(W1, W2, W1h, W2h);
    roi_align_kernel<<<MROWS, 256, 0, stream>>>(fmap, bboxes, frame_w, frame_h, A);

    dim3 g1(MROWS / 128, D1 / 128);   // 196 x 8
    wmma_gemm_relu_kernel<D1, D0, true>
        <<<g1, 256, 0, stream>>>(A, W1h, b1, (void*)H1);

    dim3 g2(MROWS / 128, D2 / 128);   // 196 x 4
    wmma_gemm_relu_kernel<D2, D1, false>
        <<<g2, 256, 0, stream>>>(H1, W2h, b2, (void*)H2);

    bin_mean_kernel<<<NROI, 256, 0, stream>>>(H2, feat);
    pair_gather_kernel<<<B * NREL, 256, 0, stream>>>(num_obj, obj_pairs, feat, out);
}
